// eSCNMDBackbone_47966194762424
// MI455X (gfx1250) — matrix operand order repaired and checked by reading, not compiled, not run on hardware
//
#include <hip/hip_runtime.h>

// ---------------- reference constants ----------------
#define CCH 128          // C
#define HCH 128          // H
#define FCH 256          // F
#define NBASIS 512      // NB
#define DIN 768          // NB + 2*C
#define SPH 9            // (LMAX+1)^2
#define XROW (SPH*CCH)   // 1152

// ---------------- WMMA types ----------------
typedef __attribute__((ext_vector_type(16))) __bf16          v16bf;
typedef __attribute__((ext_vector_type(16))) unsigned short  v16u;
typedef __attribute__((ext_vector_type(8)))  float           v8f;

// 32-byte fragment viewed as two 16B quads (-> ds_load_b128) or bf16 vector
union frag { uint4 q[2]; v16u u; v16bf h; };

__device__ __forceinline__ unsigned short f2bf(float f) {
  unsigned int b = __float_as_uint(f);
  b += 0x7FFFu + ((b >> 16) & 1u);          // round-to-nearest-even
  return (unsigned short)(b >> 16);
}
__device__ __forceinline__ float bf2f(unsigned short u) {
  return __uint_as_float(((unsigned int)u) << 16);
}
__device__ __forceinline__ float siluf(float v) { return v * (1.f / (1.f + __expf(-v))); }
__device__ __forceinline__ float sigmf(float v) { return 1.f / (1.f + __expf(-v)); }

// low 32 bits of a flat pointer to __shared__ memory == LDS byte offset
__device__ __forceinline__ unsigned lds_off(const void* p) {
  return (unsigned)(size_t)p;
}

// =====================================================================
// bf16 WMMA GEMM:  C[M x N] = epi(A[M x K] @ B[K x N])
//   A: bf16 row-major, row stride lda (16B-aligned rows).
//   B: bf16 pre-swizzled fragment-major [K/32][N/16][lane:32][16].
//   K % 32 == 0, N % 64 == 0; M arbitrary (guarded, with block-uniform
//   fast path when the whole 128-row tile is in bounds).
//   Block = 256 threads = 8 waves; tile 128(M) x 64(N); K-step 32.
//   Double-buffered LDS staging via GLOBAL_LOAD_ASYNC_TO_LDS_B128:
//   tile k+1 streams in while tile k feeds the WMMAs.
// =====================================================================
#define TM 128
#define TN 64
#define TK 32
#define APAD 8   // pad A LDS rows to 80B to spread banks

template <int EPI, int OUTBF>
__global__ __launch_bounds__(256)
void k_wmma_gemm(const unsigned short* __restrict__ A, int lda,
                 const unsigned short* __restrict__ Bsw,
                 void* __restrict__ Cp, int M, int N, int K)
{
  __shared__ unsigned short As[2][TM][TK + APAD];  // 2 x 10 KB
  __shared__ unsigned short Bs[2][4 * 32 * 16];    // 2 x 4 KB, fragment-major
  const int tid  = threadIdx.x;
  const int wv   = tid >> 5;
  const int lane = tid & 31;
  const int m0   = blockIdx.x * TM;
  const int n0   = blockIdx.y * TN;
  const int nt16 = N >> 4;
  const bool full = (m0 + TM) <= M;   // block-uniform: no row guard needed

  v8f acc[4];
  for (int t = 0; t < 4; ++t)
    for (int j = 0; j < 8; ++j) acc[t][j] = 0.f;

  auto stage = [&](int buf, int k0) {
    if (full) {
#pragma unroll
      for (int u = 0; u < 2; ++u) {              // TM*4/256 == 2 xfers/thread
        int idx = tid + u * 256;
        int r = idx >> 2, qi = idx & 3;
        const unsigned short* gp = A + (size_t)(m0 + r) * lda + k0 + qi * 8;
        unsigned lo = lds_off(&As[buf][r][qi * 8]);
        asm volatile("global_load_async_to_lds_b128 %0, %1, off"
                     :: "v"(lo), "v"(gp) : "memory");
      }
    } else {
#pragma unroll
      for (int u = 0; u < 2; ++u) {
        int idx = tid + u * 256;
        int r = idx >> 2, qi = idx & 3;
        int gr = m0 + r;
        unsigned lo = lds_off(&As[buf][r][qi * 8]);
        if (gr < M) {
          const unsigned short* gp = A + (size_t)gr * lda + k0 + qi * 8;
          asm volatile("global_load_async_to_lds_b128 %0, %1, off"
                       :: "v"(lo), "v"(gp) : "memory");
        } else {
          *(uint4*)(&As[buf][r][qi * 8]) = make_uint4(0u, 0u, 0u, 0u);
        }
      }
    }
    const unsigned short* gp =
        Bsw + (((size_t)(k0 >> 5) * nt16 + (n0 >> 4)) << 9) + tid * 8;
    unsigned lo = lds_off(&Bs[buf][tid * 8]);
    asm volatile("global_load_async_to_lds_b128 %0, %1, off"
                 :: "v"(lo), "v"(gp) : "memory");
  };

  const int kTiles = K / TK;
  stage(0, 0);                                   // prologue: tile 0 -> buf 0
  for (int kt = 0; kt < kTiles; ++kt) {
    const int buf = kt & 1;
    asm volatile("s_wait_asynccnt 0x0" ::: "memory");  // our xfers into buf done
    __syncthreads();                                    // everyone's done
    if (kt + 1 < kTiles) stage(buf ^ 1, (kt + 1) * TK); // overlap next tile

    // fragments: wide ds_load_b128 pairs, grouped ahead of the WMMAs
    frag a;
    {
      int row = wv * 16 + (lane & 15);
      int kg  = (lane >> 4) * 16;
      a.q[0] = *(const uint4*)(&As[buf][row][kg]);
      a.q[1] = *(const uint4*)(&As[buf][row][kg + 8]);
    }
    frag bfr[4];
#pragma unroll
    for (int t = 0; t < 4; ++t) {
      const unsigned short* bp = &Bs[buf][(t * 32 + lane) * 16];
      bfr[t].q[0] = *(const uint4*)(bp);
      bfr[t].q[1] = *(const uint4*)(bp + 8);
    }
    __builtin_amdgcn_sched_barrier(0);   // keep loads before the WMMA group
#pragma unroll
    for (int t = 0; t < 4; ++t) {
      acc[t] = __builtin_amdgcn_wmma_f32_16x16x32_bf16(
          false, a.h, false, bfr[t].h, (short)0, acc[t], false, false);
    }
  }

  // C/D layout: VGPR r, lanes<16 -> M=r, lanes>=16 -> M=r+8; N = lane&15
  int rb = m0 + wv * 16 + ((lane >> 4) << 3);
  int cb = n0 + (lane & 15);
#pragma unroll
  for (int t = 0; t < 4; ++t) {
#pragma unroll
    for (int r = 0; r < 8; ++r) {
      int gr = rb + r;
      if (gr < M) {
        float v = acc[t][r];
        if (EPI == 1) v = siluf(v);
        size_t o = (size_t)gr * N + cb + t * 16;
        if (OUTBF) ((unsigned short*)Cp)[o] = f2bf(v);
        else       ((float*)Cp)[o] = v;
      }
    }
  }
}

// weight swizzle: f32 [K x N] row-major -> bf16 fragment-major
// out[((kt*(N/16)+nt)*32 + lane)*16 + j] = in[(kt*32 + (lane>>4)*16 + j)*N + nt*16 + (lane&15)]
__global__ void k_swz_b(const float* __restrict__ in, unsigned short* __restrict__ out,
                        int K, int N)
{
  size_t total = (size_t)K * N;
  size_t i  = (size_t)blockIdx.x * blockDim.x + threadIdx.x;
  size_t st = (size_t)gridDim.x * blockDim.x;
  int nt16 = N >> 4;
  for (; i < total; i += st) {
    int j    = (int)(i & 15);
    int lane = (int)((i >> 4) & 31);
    size_t rest = i >> 9;
    int nt = (int)(rest % nt16);
    int kt = (int)(rest / nt16);
    int k = kt * 32 + ((lane >> 4) << 4) + j;
    int n = (nt << 4) + (lane & 15);
    out[i] = f2bf(in[(size_t)k * N + n]);
  }
}

// =====================================================================
// Per-edge Wigner construction helpers
// =====================================================================
__device__ __forceinline__ void m3mul(const float* A, const float* B, float* D) {
#pragma unroll
  for (int i = 0; i < 3; ++i)
#pragma unroll
    for (int j = 0; j < 3; ++j) {
      float s = 0.f;
#pragma unroll
      for (int k = 0; k < 3; ++k) s += A[i * 3 + k] * B[k * 3 + j];
      D[i * 3 + j] = s;
    }
}
__device__ __forceinline__ void m5mul(const float* A, const float* B, float* D) {
  for (int i = 0; i < 5; ++i)
    for (int j = 0; j < 5; ++j) {
      float s = 0.f;
      for (int k = 0; k < 5; ++k) s += A[i * 5 + k] * B[k * 5 + j];
      D[i * 5 + j] = s;
    }
}
__device__ __forceinline__ void zrot1m(float a, float* M) {
  float c = cosf(a), s = sinf(a);
  M[0] = c;   M[1] = 0.f; M[2] = s;
  M[3] = 0.f; M[4] = 1.f; M[5] = 0.f;
  M[6] = -s;  M[7] = 0.f; M[8] = c;
}
__device__ __forceinline__ void zrot2m(float a, float* M) {
  float c1 = cosf(a), s1 = sinf(a), c2 = cosf(2.f * a), s2 = sinf(2.f * a);
  for (int i = 0; i < 25; ++i) M[i] = 0.f;
  M[0]  = c2;  M[4]  = s2;
  M[6]  = c1;  M[8]  = s1;
  M[12] = 1.f;
  M[16] = -s1; M[18] = c1;
  M[20] = -s2; M[24] = c2;
}

__global__ __launch_bounds__(256)
void k_edge_geom(const float* __restrict__ pos, const float* __restrict__ rnd,
                 const int* __restrict__ src, const int* __restrict__ tgt,
                 const float* __restrict__ Jd1, const float* __restrict__ Jd2,
                 float* __restrict__ wig, float* __restrict__ dist, int E)
{
  int e = blockIdx.x * blockDim.x + threadIdx.x;
  if (e >= E) return;
  int s = src[e], t = tgt[e];
  float vx = pos[t * 3 + 0] - pos[s * 3 + 0];
  float vy = pos[t * 3 + 1] - pos[s * 3 + 1];
  float vz = pos[t * 3 + 2] - pos[s * 3 + 2];
  float dd = vx * vx + vy * vy + vz * vz;
  dist[e] = sqrtf(dd + 1e-8f);
  float inv = rsqrtf(dd + 1e-12f);
  float nx0 = vx * inv, nx1 = vy * inv, nx2 = vz * inv;

  float r0 = rnd[e * 3 + 0] - 0.5f, r1 = rnd[e * 3 + 1] - 0.5f, r2 = rnd[e * 3 + 2] - 0.5f;
  float rin = rsqrtf(r0 * r0 + r1 * r1 + r2 * r2);
  r0 *= rin; r1 *= rin; r2 *= rin;
  float b0 = -r1, b1 = r0, b2 = r2;
  float dt = fabsf(r0 * nx0 + r1 * nx1 + r2 * nx2);
  if (dt > 0.99f) { r0 = b0; r1 = b1; r2 = b2; }

  float z0 = nx1 * r2 - nx2 * r1;
  float z1 = nx2 * r0 - nx0 * r2;
  float z2 = nx0 * r1 - nx1 * r0;
  float zin = rsqrtf(z0 * z0 + z1 * z1 + z2 * z2);
  z0 *= zin; z1 *= zin; z2 *= zin;
  float y0 = nx1 * z2 - nx2 * z1;
  float y1 = nx2 * z0 - nx0 * z2;
  float y2 = nx0 * z1 - nx1 * z0;
  float yin = rsqrtf(y0 * y0 + y1 * y1 + y2 * y2);
  y0 *= yin; y1 *= yin; y2 *= yin;
  float rot[9] = { z0, z1, z2, nx0, nx1, nx2, y0, y1, y2 };

  float xc0 = rot[1], xc1 = rot[4], xc2 = rot[7];
  float xin = rsqrtf(xc0 * xc0 + xc1 * xc1 + xc2 * xc2);
  xc0 *= xin; xc1 *= xin; xc2 *= xin;
  float cbv   = fminf(fmaxf(xc1, -1.f + 1e-6f), 1.f - 1e-6f);
  float beta  = acosf(cbv);
  float alpha = atan2f(xc0, xc2);

  float ca = cosf(alpha), sa = sinf(alpha), cb = cosf(beta), sb = sinf(beta);
  float Ry[9] = { ca, 0, sa, 0, 1, 0, -sa, 0, ca };
  float Rx[9] = { 1, 0, 0, 0, cb, -sb, 0, sb, cb };
  float Ra[9]; m3mul(Ry, Rx, Ra);
  float R00 = 0.f, R02 = 0.f;               // R = Ra^T @ rot, row 0
  for (int k = 0; k < 3; ++k) { R00 += Ra[k * 3 + 0] * rot[k * 3 + 0];
                                R02 += Ra[k * 3 + 0] * rot[k * 3 + 2]; }
  float gamma = atan2f(R02, R00);

  float Z[9], T1[9], T2[9], D1[9], J1[9];
  for (int i = 0; i < 9; ++i) J1[i] = Jd1[i];
  zrot1m(alpha, Z); m3mul(Z, J1, T1);
  zrot1m(beta,  Z); m3mul(T1, Z, T2);
  m3mul(T2, J1, T1);
  zrot1m(gamma, Z); m3mul(T1, Z, D1);
  float Z5[25], U1[25], U2[25], D2[25], J2[25];
  for (int i = 0; i < 25; ++i) J2[i] = Jd2[i];
  zrot2m(alpha, Z5); m5mul(Z5, J2, U1);
  zrot2m(beta,  Z5); m5mul(U1, Z5, U2);
  m5mul(U2, J2, U1);
  zrot2m(gamma, Z5); m5mul(U1, Z5, D2);

  float* W = wig + (size_t)e * 81;
  for (int i = 0; i < 81; ++i) W[i] = 0.f;
  W[0] = 1.f;
  for (int i = 0; i < 3; ++i) for (int j = 0; j < 3; ++j) W[(1 + i) * 9 + (1 + j)] = D1[i * 3 + j];
  for (int i = 0; i < 5; ++i) for (int j = 0; j < 5; ++j) W[(4 + i) * 9 + (4 + j)] = D2[i * 5 + j];
}

__global__ __launch_bounds__(256)
void k_build_xedge(const float* __restrict__ dist, const int* __restrict__ src,
                   const int* __restrict__ tgt, const int* __restrict__ an,
                   const float* __restrict__ src_emb, const float* __restrict__ tgt_emb,
                   unsigned short* __restrict__ xedge)
{
  int e = blockIdx.x;
  float ds = dist[e];
  int as_ = an[src[e]], at = an[tgt[e]];
  const float step = 5.0f / 511.0f;
  const float w    = 2.0f * step;
  const float coef = -0.5f / (w * w);
  unsigned short* xe = xedge + (size_t)e * DIN;
  for (int idx = threadIdx.x; idx < DIN; idx += 256) {
    float v;
    if (idx < NBASIS)          { float dx = ds - step * idx; v = __expf(coef * dx * dx); }
    else if (idx < NBASIS+CCH)   v = src_emb[as_ * CCH + idx - NBASIS];
    else                         v = tgt_emb[at  * CCH + idx - NBASIS - CCH];
    xe[idx] = f2bf(v);
  }
}

__global__ __launch_bounds__(128)
void k_csd(const float* __restrict__ chg_emb, const float* __restrict__ spin_emb,
           const float* __restrict__ ds_emb, const int* __restrict__ charge,
           const int* __restrict__ spin, const int* __restrict__ dataset,
           const float* __restrict__ W_mix, const float* __restrict__ b_mix,
           float* __restrict__ csd)
{
  int b = blockIdx.x, c = threadIdx.x;
  __shared__ float vin[3 * CCH];
  for (int idx = c; idx < 3 * CCH; idx += 128) {
    float v;
    if (idx < CCH)           v = chg_emb [charge [b] * CCH + idx];
    else if (idx < 2 * CCH)  v = spin_emb[spin   [b] * CCH + idx - CCH];
    else                     v = ds_emb  [dataset[b] * CCH + idx - 2 * CCH];
    vin[idx] = v;
  }
  __syncthreads();
  float a = b_mix[c];
  for (int k = 0; k < 3 * CCH; ++k) a += vin[k] * W_mix[k * CCH + c];
  csd[b * CCH + c] = siluf(a);
}

__global__ __launch_bounds__(128)
void k_node_init(const float* __restrict__ sphere_emb, const float* __restrict__ csd,
                 const int* __restrict__ an, const int* __restrict__ batch,
                 float* __restrict__ x)
{
  int n = blockIdx.x, c = threadIdx.x;
  float* xb = x + (size_t)n * XROW;
  xb[c] = sphere_emb[an[n] * CCH + c] + csd[batch[n] * CCH + c];
  for (int i = 1; i < SPH; ++i) xb[i * CCH + c] = 0.f;
}

__global__ __launch_bounds__(128)
void k_embed_scatter(const unsigned short* __restrict__ m0, const float* __restrict__ wig,
                     const int* __restrict__ tgt, float* __restrict__ x)
{
  int e = blockIdx.x, c = threadIdx.x;
  __shared__ float w[81];
  if (c < 81) w[c] = wig[(size_t)e * 81 + c];
  float a0 = bf2f(m0[(size_t)e * 384 + c]);
  float a1 = bf2f(m0[(size_t)e * 384 + 128 + c]);
  float a2 = bf2f(m0[(size_t)e * 384 + 256 + c]);
  __syncthreads();
  float* xb = x + (size_t)tgt[e] * XROW;
#pragma unroll
  for (int i = 0; i < SPH; ++i) {
    float v = w[0 * 9 + i] * a0 + w[2 * 9 + i] * a1 + w[6 * 9 + i] * a2;
    atomicAdd(&xb[i * CCH + c], 0.2f * v);
  }
}

__global__ __launch_bounds__(128)
void k_msg_rotate(const float* __restrict__ xinf, const float* __restrict__ wig,
                  const int* __restrict__ src, const int* __restrict__ tgt,
                  unsigned short* __restrict__ msg)
{
  int e = blockIdx.x, c = threadIdx.x;
  __shared__ float w[81];
  if (c < 81) w[c] = wig[(size_t)e * 81 + c];
  const float* xs = xinf + (size_t)src[e] * XROW;
  const float* xt = xinf + (size_t)tgt[e] * XROW;
  float sv[9];
#pragma unroll
  for (int j = 0; j < 9; ++j) sv[j] = xs[j * CCH + c] + xt[j * CCH + c];
  __syncthreads();
  unsigned short* mb = msg + (size_t)e * SPH * CCH;
#pragma unroll
  for (int i = 0; i < 9; ++i) {
    float a = 0.f;
#pragma unroll
    for (int j = 0; j < 9; ++j) a += w[i * 9 + j] * sv[j];
    mb[i * CCH + c] = f2bf(a);
  }
}

__global__ __launch_bounds__(128)
void k_gate_msg(const unsigned short* __restrict__ hmraw, const float* __restrict__ rad,
                unsigned short* __restrict__ hmg)
{
  int e = blockIdx.x, c = threadIdx.x;
  float rv = rad[(size_t)e * CCH + c];
  const unsigned short* hb = hmraw + (size_t)e * SPH * CCH;
  unsigned short*       ob = hmg   + (size_t)e * SPH * CCH;
  float h0 = bf2f(hb[c]) * rv;
  float g  = sigmf(h0);
  ob[c] = f2bf(siluf(h0));
#pragma unroll
  for (int i = 1; i < SPH; ++i)
    ob[i * CCH + c] = f2bf(bf2f(hb[i * CCH + c]) * rv * g);
}

__global__ __launch_bounds__(128)
void k_rot_back_scatter(const unsigned short* __restrict__ hm2, const float* __restrict__ wig,
                        const int* __restrict__ tgt, float* __restrict__ x)
{
  int e = blockIdx.x, c = threadIdx.x;
  __shared__ float w[81];
  if (c < 81) w[c] = wig[(size_t)e * 81 + c];
  const unsigned short* hb = hm2 + (size_t)e * SPH * CCH;
  float m[9];
#pragma unroll
  for (int j = 0; j < 9; ++j) m[j] = bf2f(hb[j * CCH + c]);
  __syncthreads();
  float* xb = x + (size_t)tgt[e] * XROW;
#pragma unroll
  for (int i = 0; i < SPH; ++i) {
    float a = 0.f;
#pragma unroll
    for (int j = 0; j < 9; ++j) a += w[j * 9 + i] * m[j];
    atomicAdd(&xb[i * CCH + c], a);
  }
}

__global__ __launch_bounds__(256)
void k_gate_ff(const unsigned short* __restrict__ hf, const float* __restrict__ gateraw,
               unsigned short* __restrict__ hfg)
{
  int n = blockIdx.x, h = threadIdx.x;
  float g = sigmf(gateraw[(size_t)n * FCH + h]);
  const unsigned short* hb = hf  + (size_t)n * SPH * FCH;
  unsigned short*       ob = hfg + (size_t)n * SPH * FCH;
  ob[h] = f2bf(siluf(bf2f(hb[h])));
#pragma unroll
  for (int i = 1; i < SPH; ++i)
    ob[i * FCH + h] = f2bf(bf2f(hb[i * FCH + h]) * g);
}

__global__ __launch_bounds__(128)
void k_rmsnorm(const float* __restrict__ x, const float* __restrict__ w,
               float* __restrict__ outf, unsigned short* __restrict__ outb)
{
  int n = blockIdx.x, c = threadIdx.x;
  __shared__ float red[128];
  const int segA[3] = { 0, 1, 4 };
  const int segB[3] = { 1, 4, 9 };
  const float* xb = x + (size_t)n * XROW;
  float* of = outf + (size_t)n * XROW;
  for (int li = 0; li < 3; ++li) {
    int a = segA[li], b = segB[li], d = b - a;
    float ss = 0.f;
    for (int i = a; i < b; ++i) { float v = xb[i * CCH + c]; ss += v * v; }
    red[c] = ss; __syncthreads();
    for (int off = 64; off > 0; off >>= 1) { if (c < off) red[c] += red[c + off]; __syncthreads(); }
    float tot = red[0];
    __syncthreads();
    float scale = w[li * CCH + c] * rsqrtf(tot / (float)(d * CCH) + 1e-6f);
    for (int i = a; i < b; ++i) {
      float v = xb[i * CCH + c] * scale;
      of[i * CCH + c] = v;
      if (outb) outb[(size_t)n * XROW + i * CCH + c] = f2bf(v);
    }
  }
}

__global__ void k_add(float* __restrict__ x, const float* __restrict__ y, size_t n) {
  size_t i = (size_t)blockIdx.x * blockDim.x + threadIdx.x;
  size_t st = (size_t)gridDim.x * blockDim.x;
  for (; i < n; i += st) x[i] += y[i];
}

// =====================================================================
// host orchestration
// =====================================================================
static inline size_t al256(size_t b) { return (b + 255) & ~(size_t)255; }

extern "C" void kernel_launch(void* const* d_in, const int* in_sizes, int n_in,
                              void* d_out, int out_size, void* d_ws, size_t ws_size,
                              hipStream_t stream)
{
  (void)n_in; (void)out_size; (void)ws_size;
  const float* pos        = (const float*)d_in[0];
  const float* edge_rand  = (const float*)d_in[1];
  const float* sphere_emb = (const float*)d_in[2];
  const float* chg_emb    = (const float*)d_in[3];
  const float* spin_emb   = (const float*)d_in[4];
  const float* ds_emb     = (const float*)d_in[5];
  const float* W_mix      = (const float*)d_in[6];
  const float* b_mix      = (const float*)d_in[7];
  const float* src_emb    = (const float*)d_in[8];
  const float* tgt_emb    = (const float*)d_in[9];
  const float* Wdeg1      = (const float*)d_in[10];
  const float* Wdeg2      = (const float*)d_in[11];
  const float* Wdeg3      = (const float*)d_in[12];
  const float* Jd1        = (const float*)d_in[13];
  const float* Jd2        = (const float*)d_in[14];
  const float* Wrad1      = (const float*)d_in[15];
  const float* Wrad2      = (const float*)d_in[16];
  const float* Wmsg_in    = (const float*)d_in[17];
  const float* Wmsg_out   = (const float*)d_in[18];
  const float* Wff1       = (const float*)d_in[19];
  const float* Wgate      = (const float*)d_in[20];
  const float* Wff2       = (const float*)d_in[21];
  const float* w_norm1    = (const float*)d_in[22];
  const float* w_norm2    = (const float*)d_in[23];
  const float* w_normf    = (const float*)d_in[24];
  const int*   an         = (const int*)d_in[25];
  const int*   edge_index = (const int*)d_in[26];
  const int*   batch      = (const int*)d_in[27];
  const int*   charge     = (const int*)d_in[28];
  const int*   spin       = (const int*)d_in[29];
  const int*   dataset    = (const int*)d_in[30];

  const int N   = in_sizes[0] / 3;
  const int E   = in_sizes[1] / 3;
  const int B   = in_sizes[28];
  const int NBL = in_sizes[15] / (DIN * HCH);
  const int* src = edge_index;
  const int* tgt = edge_index + E;

  // -------- workspace arena --------
  char* p = (char*)d_ws;
  auto alloc = [&](size_t bytes) -> char* { char* r = p; p += al256(bytes); return r; };

  unsigned short* wdeg1b = (unsigned short*)alloc((size_t)DIN * CCH * 2);
  unsigned short* wdeg2b = (unsigned short*)alloc((size_t)CCH * CCH * 2);
  unsigned short* wdeg3b = (unsigned short*)alloc((size_t)CCH * 3 * CCH * 2);
  unsigned short* wrad1b = (unsigned short*)alloc((size_t)NBL * DIN * HCH * 2);
  unsigned short* wrad2b = (unsigned short*)alloc((size_t)NBL * HCH * HCH * 2);
  unsigned short* wmib   = (unsigned short*)alloc((size_t)NBL * CCH * HCH * 2);
  unsigned short* wmob   = (unsigned short*)alloc((size_t)NBL * HCH * CCH * 2);
  unsigned short* wff1b  = (unsigned short*)alloc((size_t)NBL * CCH * FCH * 2);
  unsigned short* wgateb = (unsigned short*)alloc((size_t)NBL * CCH * FCH * 2);
  unsigned short* wff2b  = (unsigned short*)alloc((size_t)NBL * FCH * CCH * 2);

  float*          wig   = (float*)alloc((size_t)E * 81 * 4);
  float*          dist  = (float*)alloc((size_t)E * 4);
  float*          csd   = (float*)alloc((size_t)B * CCH * 4);
  float*          x     = (float*)alloc((size_t)N * XROW * 4);
  unsigned short* xedge = (unsigned short*)alloc((size_t)E * DIN * 2);
  float*          xinf  = (float*)alloc((size_t)N * XROW * 4);
  unsigned short* xinb  = (unsigned short*)alloc((size_t)N * XROW * 2);
  unsigned short* radt  = (unsigned short*)alloc((size_t)E * HCH * 2);
  float*          radf  = (float*)alloc((size_t)E * HCH * 4);
  unsigned short* X1u   = (unsigned short*)alloc((size_t)E * SPH * CCH * 2);
  unsigned short* X2u   = (unsigned short*)alloc((size_t)E * SPH * CCH * 2);

  float*          gateraw = (float*)X1u;
  unsigned short* hfg     = (unsigned short*)((char*)X1u + al256((size_t)N * FCH * 4));
  float*          ffout   = (float*)((char*)hfg + al256((size_t)N * SPH * FCH * 2));

  auto gemm = [&](const void* A, int lda, const void* Bm, void* Cm,
                  int M, int Ncol, int K, int epi, int outbf) {
    dim3 g((M + TM - 1) / TM, Ncol / TN), blk(256);
    const unsigned short* Ap = (const unsigned short*)A;
    const unsigned short* Bp = (const unsigned short*)Bm;
    if      (epi == 0 && outbf == 0) k_wmma_gemm<0,0><<<g, blk, 0, stream>>>(Ap, lda, Bp, Cm, M, Ncol, K);
    else if (epi == 0 && outbf == 1) k_wmma_gemm<0,1><<<g, blk, 0, stream>>>(Ap, lda, Bp, Cm, M, Ncol, K);
    else if (epi == 1 && outbf == 0) k_wmma_gemm<1,0><<<g, blk, 0, stream>>>(Ap, lda, Bp, Cm, M, Ncol, K);
    else                             k_wmma_gemm<1,1><<<g, blk, 0, stream>>>(Ap, lda, Bp, Cm, M, Ncol, K);
  };
  auto swzB = [&](const float* in, unsigned short* out, int K, int Ncol, int L) {
    for (int i = 0; i < L; ++i)
      k_swz_b<<<256, 256, 0, stream>>>(in + (size_t)i * K * Ncol,
                                       out + (size_t)i * K * Ncol, K, Ncol);
  };

  // -------- weight conversion + swizzle to fragment-major --------
  swzB(Wdeg1,    wdeg1b, DIN, CCH,     1);
  swzB(Wdeg2,    wdeg2b, CCH, CCH,     1);
  swzB(Wdeg3,    wdeg3b, CCH, 3 * CCH, 1);
  swzB(Wrad1,    wrad1b, DIN, HCH,   NBL);
  swzB(Wrad2,    wrad2b, HCH, HCH,   NBL);
  swzB(Wmsg_in,  wmib,   CCH, HCH,   NBL);
  swzB(Wmsg_out, wmob,   HCH, CCH,   NBL);
  swzB(Wff1,     wff1b,  CCH, FCH,   NBL);
  swzB(Wgate,    wgateb, CCH, FCH,   NBL);
  swzB(Wff2,     wff2b,  FCH, CCH,   NBL);

  // -------- node init + edge geometry + edge features --------
  k_csd<<<B, 128, 0, stream>>>(chg_emb, spin_emb, ds_emb, charge, spin, dataset, W_mix, b_mix, csd);
  k_node_init<<<N, 128, 0, stream>>>(sphere_emb, csd, an, batch, x);
  k_edge_geom<<<(E + 255) / 256, 256, 0, stream>>>(pos, edge_rand, src, tgt, Jd1, Jd2, wig, dist, E);
  k_build_xedge<<<E, 256, 0, stream>>>(dist, src, tgt, an, src_emb, tgt_emb, xedge);

  // -------- initial edge-degree embedding (WMMA chain) --------
  gemm(xedge, DIN, wdeg1b, X1u, E, CCH,     DIN, 1, 1);
  gemm(X1u,   CCH, wdeg2b, X2u, E, CCH,     CCH, 1, 1);
  gemm(X2u,   CCH, wdeg3b, X1u, E, 3 * CCH, CCH, 0, 1);
  k_embed_scatter<<<E, 128, 0, stream>>>(X1u, wig, tgt, x);

  // -------- message-passing layers --------
  for (int i = 0; i < NBL; ++i) {
    k_rmsnorm<<<N, 128, 0, stream>>>(x, w_norm1 + (size_t)i * 3 * CCH, xinf, xinb);
    k_msg_rotate<<<E, 128, 0, stream>>>(xinf, wig, src, tgt, X1u);
    gemm(xedge, DIN, wrad1b + (size_t)i * DIN * HCH, radt, E, HCH, DIN, 1, 1);
    gemm(radt,  HCH, wrad2b + (size_t)i * HCH * HCH, radf, E, HCH, HCH, 1, 0);
    gemm(X1u,   CCH, wmib   + (size_t)i * CCH * HCH, X2u, E * SPH, HCH, CCH, 0, 1);
    k_gate_msg<<<E, 128, 0, stream>>>(X2u, radf, X1u);
    gemm(X1u,   HCH, wmob   + (size_t)i * HCH * CCH, X2u, E * SPH, CCH, HCH, 0, 1);
    k_rot_back_scatter<<<E, 128, 0, stream>>>(X2u, wig, tgt, x);

    k_rmsnorm<<<N, 128, 0, stream>>>(x, w_norm2 + (size_t)i * 3 * CCH, xinf, xinb);
    gemm(xinb, CCH,  wff1b  + (size_t)i * CCH * FCH, X2u,     N * SPH, FCH, CCH, 0, 1);
    gemm(xinb, XROW, wgateb + (size_t)i * CCH * FCH, gateraw, N,       FCH, CCH, 0, 0);
    k_gate_ff<<<N, 256, 0, stream>>>(X2u, gateraw, hfg);
    gemm(hfg,  FCH,  wff2b  + (size_t)i * FCH * CCH, ffout,   N * SPH, CCH, FCH, 0, 0);
    k_add<<<2048, 256, 0, stream>>>(x, ffout, (size_t)N * XROW);
  }

  // -------- final norm -> f32 output --------
  k_rmsnorm<<<N, 128, 0, stream>>>(x, w_normf, (float*)d_out, nullptr);
}
